// TransformerEncoderLayer_62491774157580
// MI455X (gfx1250) — matrix operand order repaired and checked by reading, not compiled, not run on hardware
//
#include <hip/hip_runtime.h>
#include <math.h>
#include <stdint.h>

typedef _Float16 h16;
typedef __attribute__((ext_vector_type(16))) _Float16 v16h;
typedef __attribute__((ext_vector_type(8)))  _Float16 v8h;
typedef __attribute__((ext_vector_type(8)))  float    v8f;

// ---------------------------------------------------------------------------
// Problem constants
// ---------------------------------------------------------------------------
#define NTB   128          // NB*TT = 8*16
#define LTOK  197
#define CCH   768
#define DQKV  2304         // fused q|k|v output width
#define NHEAD 12
#define HDIM  64
#define MMLP  3072
#define RDIM  512
#define TT_   16
#define HWTOK 196
#define ROWS_T (NTB*LTOK)   // 25216 = 197*128
#define ROWS_S (NTB*HWTOK)  // 25088 = 196*128
#define LPAD  224           // kv padded to 14*16 for attention

// ---------------------------------------------------------------------------
// WMMA fragment loaders (16x16x32 f16, contiguous k order, same for A and B)
// A 16x32: lanes 0-15 -> row=lane, K 0..15 contiguous; lanes 16-31 -> K 16..31
// B 32x16: lanes 0-15 -> col=lane, K 0..15 contiguous; lanes 16-31 -> K 16..31
// ---------------------------------------------------------------------------
__device__ __forceinline__ v16h frag_a(const h16* base, int ld) {
  int lane = threadIdx.x & 31;
  int row  = lane & 15;
  int hi   = lane >> 4;
  const h16* p = base + (size_t)row * ld + hi * 16;
  union { v16h v; v8h h[2]; } u;
  u.h[0] = *(const v8h*)(p);
  u.h[1] = *(const v8h*)(p + 8);
  return u.v;
}
__device__ __forceinline__ v16h frag_b(const h16* base, int ld) {
  int lane = threadIdx.x & 31;
  int col  = lane & 15;
  int hi   = lane >> 4;
  return *(const v16h*)(base + (size_t)col * ld + hi * 16);
}
#define WMMA_F32_F16(a, b, c) \
  __builtin_amdgcn_wmma_f32_16x16x32_f16(false, (a), false, (b), (short)0, (c), false, false)

// ---------------------------------------------------------------------------
// Generic WMMA GEMM:  out[M,N] = A[M,K] (f16) * W[N,K]^T (f16) + bias
// Block = 4 waves; block tile 128M x 64N; wave tile 32M x 64N (8 WMMA / K-step).
// The 64x32 B slice is staged once per block into LDS with
// global_load_async_to_lds_b128, double-buffered on ASYNCcnt.
// MODE 0: f16 out + bias                  (fused QKV)
// MODE 1: f16 out, qgelu(bias+acc)        (MLP fc1)
// MODE 2: f32 out + bias                  (temporal c1)
// MODE 3: f32 out = resid[xmap] + acc + b (temporal c2; spatial->token remap)
// MODE 4: f32 out = resid + acc + b       (O-proj, fc2)
// Requires M % 128 == 0, N % 64 == 0, K % 32 == 0 (all true here).
// ---------------------------------------------------------------------------
template <int MODE>
__global__ __launch_bounds__(128)
void wmma_gemm(const h16* __restrict__ A, const h16* __restrict__ W,
               const float* __restrict__ bias,
               float* __restrict__ outF, h16* __restrict__ outH,
               const float* __restrict__ resid,
               int M, int N, int K)
{
  __shared__ __attribute__((aligned(128))) h16 Bs[2][64 * 32];

  int tid  = (int)threadIdx.x;
  int wave = tid >> 5;
  int m0 = (int)blockIdx.x * 128 + wave * 32;
  int n0 = (int)blockIdx.y * 64;

  // Stage the 64-row x 32-half B slice for K-offset k into Bs[buf].
  // 256 x 16B chunks, 2 per thread, issued as async loads (ASYNCcnt += 2).
  const int c0 = tid * 2;
  auto stage = [&](int buf, int k) {
    uint32_t lbase = (uint32_t)(uintptr_t)(&Bs[buf][0]);   // ptrtoint escapes Bs
#pragma unroll
    for (int i = 0; i < 2; ++i) {
      int ci = c0 + i;
      int row = ci >> 2, part = ci & 3;
      const h16* g = W + (size_t)(n0 + row) * K + k + part * 8;
      uint32_t l = lbase + (uint32_t)(row * 32 + part * 8) * 2;
      asm volatile("global_load_async_to_lds_b128 %0, %1, off"
                   :: "v"(l), "v"(g) : "memory");
    }
  };

  v8f acc[8];
#pragma unroll
  for (int j = 0; j < 8; ++j) acc[j] = (v8f){0.f,0.f,0.f,0.f,0.f,0.f,0.f,0.f};

  const h16* arow0 = A + (size_t)m0 * K;
  const h16* arow1 = arow0 + (size_t)16 * K;
  int steps = K >> 5;

  stage(0, 0);
  for (int ks = 0; ks < steps; ++ks) {
    int cur = ks & 1;
    if (ks + 1 < steps) {
      stage(cur ^ 1, (ks + 1) << 5);
      asm volatile("s_wait_asynccnt 2" ::: "memory");   // cur slice landed
    } else {
      asm volatile("s_wait_asynccnt 0" ::: "memory");
    }
    __syncthreads();                                    // all waves' stages landed

    int k = ks << 5;
    v16h a0 = frag_a(arow0 + k, K);
    v16h a1 = frag_a(arow1 + k, K);
    const h16* bs = &Bs[cur][0];
    asm volatile("" : "+r"(bs));                        // launder: asm wrote LDS
#pragma unroll
    for (int j = 0; j < 4; ++j) {
      v16h b = frag_b(bs + j * 16 * 32, 32);
      acc[j]     = WMMA_F32_F16(a0, b, acc[j]);
      acc[4 + j] = WMMA_F32_F16(a1, b, acc[4 + j]);
    }
    __syncthreads();                                    // done reading cur buf
  }

  int lane = tid & 31;
  int cn = lane & 15;
  int r0 = (lane >> 4) * 8;
#pragma unroll
  for (int half = 0; half < 2; ++half) {
#pragma unroll
    for (int j = 0; j < 4; ++j) {
      int col = n0 + j * 16 + cn;
      float bb = bias[col];
      v8f a = acc[half * 4 + j];
#pragma unroll
      for (int r = 0; r < 8; ++r) {
        int row = m0 + half * 16 + r0 + r;
        float v = a[r] + bb;
        if (MODE == 0) {
          outH[(size_t)row * N + col] = (h16)v;
        } else if (MODE == 1) {
          float g = v / (1.f + __expf(-1.702f * v));   // qgelu
          outH[(size_t)row * N + col] = (h16)g;
        } else if (MODE == 2) {
          outF[(size_t)row * N + col] = v;
        } else if (MODE == 3) {
          int nt = row / HWTOK, hw = row % HWTOK;
          size_t xo = ((size_t)nt * LTOK + 1 + hw) * (size_t)N + col;
          outF[xo] = resid[xo] + v;
        } else {
          size_t xo = (size_t)row * N + col;
          outF[xo] = resid[xo] + v;
        }
      }
    }
  }
}

// ---------------------------------------------------------------------------
// Fused attention: one wave per (nt, head, 16-query tile).
// Q/K read from fused QKV buffer (row stride DQKV); V pre-transposed per head.
// scores = (Q*scale)K^T via WMMA -> masked softmax over kv in LDS -> P*V via WMMA.
// ---------------------------------------------------------------------------
__global__ __launch_bounds__(32)
void attn_kernel(const h16* __restrict__ QKV, const h16* __restrict__ VT,
                 h16* __restrict__ Mix)
{
  __shared__ __attribute__((aligned(32))) float sc[16 * LPAD];
  __shared__ __attribute__((aligned(32))) h16   ph[16 * LPAD];

  int nt = (int)blockIdx.x, head = (int)blockIdx.y, qt = (int)blockIdx.z;
  int lane = (int)threadIdx.x;
  int cn = lane & 15, r0 = (lane >> 4) * 8;
  const float scale = 0.125f;   // 1/sqrt(64)

  size_t qoff = ((size_t)nt * LTOK + qt * 16) * DQKV + head * HDIM;
  v16h a0 = frag_a(QKV + qoff,      DQKV);
  v16h a1 = frag_a(QKV + qoff + 32, DQKV);

  for (int kt = 0; kt < 13; ++kt) {
    size_t koff = ((size_t)nt * LTOK + kt * 16) * DQKV + CCH + head * HDIM;
    v16h b0 = frag_b(QKV + koff,      DQKV);
    v16h b1 = frag_b(QKV + koff + 32, DQKV);
    v8f s = (v8f){0.f,0.f,0.f,0.f,0.f,0.f,0.f,0.f};
    s = WMMA_F32_F16(a0, b0, s);
    s = WMMA_F32_F16(a1, b1, s);
    int col = kt * 16 + cn;
#pragma unroll
    for (int r = 0; r < 8; ++r)
      sc[(r0 + r) * LPAD + col] = (col < LTOK) ? s[r] * scale : -3.0e38f;
  }
  __syncthreads();

  // row-wise softmax over kv (cols 0..207 written, >=197 masked to -inf)
  for (int row = 0; row < 16; ++row) {
    float mx = -3.0e38f;
    for (int n = lane; n < 208; n += 32) mx = fmaxf(mx, sc[row * LPAD + n]);
#pragma unroll
    for (int m = 16; m > 0; m >>= 1) mx = fmaxf(mx, __shfl_xor(mx, m, 32));
    float sum = 0.f;
    for (int n = lane; n < LPAD; n += 32) {
      float e = (n < 208) ? __expf(sc[row * LPAD + n] - mx) : 0.f;
      sum += e;
      sc[row * LPAD + n] = e;
    }
#pragma unroll
    for (int m = 16; m > 0; m >>= 1) sum += __shfl_xor(sum, m, 32);
    float inv = 1.f / sum;
    for (int n = lane; n < LPAD; n += 32)
      ph[row * LPAD + n] = (h16)(sc[row * LPAD + n] * inv);
  }
  __syncthreads();

  // mix = P (16 x 224) * V (224 x 64), V pre-transposed per head (kv contiguous)
  v8f am[4];
#pragma unroll
  for (int d = 0; d < 4; ++d) am[d] = (v8f){0.f,0.f,0.f,0.f,0.f,0.f,0.f,0.f};
  const h16* vbase = VT + ((size_t)(nt * NHEAD + head)) * HDIM * LPAD;
  for (int kk = 0; kk < 7; ++kk) {
    v16h a = frag_a(ph + kk * 32, LPAD);
#pragma unroll
    for (int d = 0; d < 4; ++d) {
      v16h b = frag_b(vbase + (size_t)(d * 16) * LPAD + kk * 32, LPAD);
      am[d] = WMMA_F32_F16(a, b, am[d]);
    }
  }
#pragma unroll
  for (int d = 0; d < 4; ++d) {
    int col = head * HDIM + d * 16 + cn;
#pragma unroll
    for (int r = 0; r < 8; ++r) {
      int q = qt * 16 + r0 + r;
      if (q < LTOK)
        Mix[((size_t)nt * LTOK + q) * CCH + col] = (h16)am[d][r];
    }
  }
}

// ---------------------------------------------------------------------------
// LayerNorm: one block (256 thr) per token row, f32 in -> f16 out
// ---------------------------------------------------------------------------
__global__ __launch_bounds__(256)
void layernorm_kernel(const float* __restrict__ X, const float* __restrict__ g,
                      const float* __restrict__ b, h16* __restrict__ out)
{
  __shared__ float red[256];
  int row = (int)blockIdx.x, tid = (int)threadIdx.x;
  const float* xr = X + (size_t)row * CCH;
  float v0 = xr[tid], v1 = xr[tid + 256], v2 = xr[tid + 512];
  red[tid] = v0 + v1 + v2;
  __syncthreads();
  for (int o = 128; o > 0; o >>= 1) { if (tid < o) red[tid] += red[tid + o]; __syncthreads(); }
  float mean = red[0] * (1.f / 768.f);
  __syncthreads();
  float d0 = v0 - mean, d1 = v1 - mean, d2 = v2 - mean;
  red[tid] = d0 * d0 + d1 * d1 + d2 * d2;
  __syncthreads();
  for (int o = 128; o > 0; o >>= 1) { if (tid < o) red[tid] += red[tid + o]; __syncthreads(); }
  float inv = rsqrtf(red[0] * (1.f / 768.f) + 1e-5f);
  h16* orow = out + (size_t)row * CCH;
  orow[tid]       = (h16)(d0 * inv * g[tid]       + b[tid]);
  orow[tid + 256] = (h16)(d1 * inv * g[tid + 256] + b[tid + 256]);
  orow[tid + 512] = (h16)(d2 * inv * g[tid + 512] + b[tid + 512]);
}

// ---------------------------------------------------------------------------
// BN affine (eval) on spatial tokens: (x - rm)*g*rsqrt(rv+eps)+b, f32 -> f16
// ---------------------------------------------------------------------------
__global__ __launch_bounds__(256)
void bnaff_kernel(const float* __restrict__ X, const float* __restrict__ g,
                  const float* __restrict__ b, const float* __restrict__ rm,
                  const float* __restrict__ rv, h16* __restrict__ out)
{
  size_t idx = (size_t)blockIdx.x * 256 + threadIdx.x;
  if (idx >= (size_t)ROWS_S * CCH) return;
  int c = (int)(idx % CCH);
  size_t row = idx / CCH;
  int nt = (int)(row / HWTOK), hw = (int)(row % HWTOK);
  float xv = X[((size_t)nt * LTOK + 1 + hw) * CCH + c];
  float s = g[c] * rsqrtf(rv[c] + 1e-5f);
  out[idx] = (h16)((xv - rm[c]) * s + b[c]);
}

// ---------------------------------------------------------------------------
// Depthwise temporal conv (k=3, zero pad over T), f32 in -> f16 out
// ---------------------------------------------------------------------------
__global__ __launch_bounds__(256)
void dwconv_kernel(const float* __restrict__ H1, const float* __restrict__ w,
                   const float* __restrict__ b, h16* __restrict__ out)
{
  size_t idx = (size_t)blockIdx.x * 256 + threadIdx.x;
  if (idx >= (size_t)ROWS_S * RDIM) return;
  int r = (int)(idx % RDIM);
  size_t row = idx / RDIM;
  int hw = (int)(row % HWTOK);
  int nt = (int)(row / HWTOK);
  int t = nt % TT_, n = nt / TT_;
  float acc = b[r];
#pragma unroll
  for (int kk = 0; kk < 3; ++kk) {
    int tt = t + kk - 1;
    if (tt >= 0 && tt < TT_)
      acc += w[r * 3 + kk] * H1[(((size_t)(n * TT_ + tt)) * HWTOK + hw) * RDIM + r];
  }
  out[idx] = (h16)acc;
}

// ---------------------------------------------------------------------------
// V transpose per head from fused QKV: VT[nt][h][d][kv(224)]
// ---------------------------------------------------------------------------
__global__ __launch_bounds__(256)
void vtrans_kernel(const h16* __restrict__ QKV, h16* __restrict__ VT)
{
  size_t idx = (size_t)blockIdx.x * 256 + threadIdx.x;
  if (idx >= (size_t)NTB * NHEAD * HDIM * LPAD) return;
  int kv = (int)(idx % LPAD);
  size_t rest = idx / LPAD;
  int d = (int)(rest % HDIM); rest /= HDIM;
  int h = (int)(rest % NHEAD);
  int nt = (int)(rest / NHEAD);
  h16 val = (h16)0.f;
  if (kv < LTOK)
    val = QKV[((size_t)nt * LTOK + kv) * DQKV + 2 * CCH + h * HDIM + d];
  VT[idx] = val;
}

__global__ __launch_bounds__(256)
void castf2h_kernel(const float* __restrict__ in, h16* __restrict__ out, long n)
{
  long i = (long)blockIdx.x * 256 + threadIdx.x;
  if (i < n) out[i] = (h16)in[i];
}

__global__ __launch_bounds__(256)
void concat3_kernel(const float* __restrict__ a, const float* __restrict__ b,
                    const float* __restrict__ c, float* __restrict__ o)
{
  int i = (int)blockIdx.x * 256 + threadIdx.x;
  if (i >= DQKV) return;
  o[i] = (i < CCH) ? a[i] : (i < 2 * CCH) ? b[i - CCH] : c[i - 2 * CCH];
}

__global__ __launch_bounds__(256)
void clscopy_kernel(const float* __restrict__ X, float* __restrict__ out)
{
  int idx = (int)blockIdx.x * 256 + threadIdx.x;
  if (idx >= NTB * CCH) return;
  int nt = idx / CCH, c = idx % CCH;
  size_t o = ((size_t)nt * LTOK) * CCH + c;
  out[o] = X[o];
}

// ---------------------------------------------------------------------------
// Host-side orchestration
// ---------------------------------------------------------------------------
extern "C" void kernel_launch(void* const* d_in, const int* in_sizes, int n_in,
                              void* d_out, int out_size, void* d_ws, size_t ws_size,
                              hipStream_t stream)
{
  (void)in_sizes; (void)n_in; (void)out_size; (void)ws_size;
  const float* x     = (const float*)d_in[0];
  const float* q_w   = (const float*)d_in[1];
  const float* q_b   = (const float*)d_in[2];
  const float* k_w   = (const float*)d_in[3];
  const float* k_b   = (const float*)d_in[4];
  const float* v_w   = (const float*)d_in[5];
  const float* v_b   = (const float*)d_in[6];
  const float* o_w   = (const float*)d_in[7];
  const float* o_b   = (const float*)d_in[8];
  const float* ln1_g = (const float*)d_in[9];
  const float* ln1_b = (const float*)d_in[10];
  const float* ln2_g = (const float*)d_in[11];
  const float* ln2_b = (const float*)d_in[12];
  const float* fc1_w = (const float*)d_in[13];
  const float* fc1_b = (const float*)d_in[14];
  const float* fc2_w = (const float*)d_in[15];
  const float* fc2_b = (const float*)d_in[16];
  const float* tm_bn_g[2]  = {(const float*)d_in[17], (const float*)d_in[27]};
  const float* tm_bn_b[2]  = {(const float*)d_in[18], (const float*)d_in[28]};
  const float* tm_bn_rm[2] = {(const float*)d_in[19], (const float*)d_in[29]};
  const float* tm_bn_rv[2] = {(const float*)d_in[20], (const float*)d_in[30]};
  const float* tm_c1_w[2]  = {(const float*)d_in[21], (const float*)d_in[31]};
  const float* tm_c1_b[2]  = {(const float*)d_in[22], (const float*)d_in[32]};
  const float* tm_dw_w[2]  = {(const float*)d_in[23], (const float*)d_in[33]};
  const float* tm_dw_b[2]  = {(const float*)d_in[24], (const float*)d_in[34]};
  const float* tm_c2_w[2]  = {(const float*)d_in[25], (const float*)d_in[35]};
  const float* tm_c2_b[2]  = {(const float*)d_in[26], (const float*)d_in[36]};
  float* out = (float*)d_out;

  // ---- workspace bump allocator (256B aligned) ----
  char* ws = (char*)d_ws;
  size_t off = 0;
  auto alloc = [&](size_t bytes) -> char* {
    char* p = ws + off;
    off = (off + bytes + 255) & ~(size_t)255;
    return p;
  };
  float* x_work = (float*)alloc((size_t)ROWS_T * CCH * 4);
  h16*   xn     = (h16*)  alloc((size_t)ROWS_T * CCH * 2);
  h16*   wqkv   = (h16*)  alloc((size_t)DQKV * CCH * 2);   // q|k|v rows concatenated
  float* bqkv   = (float*)alloc((size_t)DQKV * 4);
  h16*   wo     = (h16*)  alloc((size_t)CCH * CCH * 2);
  h16*   wfc1   = (h16*)  alloc((size_t)MMLP * CCH * 2);
  h16*   wfc2   = (h16*)  alloc((size_t)CCH * MMLP * 2);
  h16*   wc1[2], *wc2[2];
  for (int i = 0; i < 2; ++i) {
    wc1[i] = (h16*)alloc((size_t)RDIM * CCH * 2);
    wc2[i] = (h16*)alloc((size_t)CCH * RDIM * 2);
  }
  // phase-overlapped scratch region (max over temporal / attention / MLP use)
  size_t padRows = ROWS_T + 16;                      // OOB-read slack for padded q-tiles
  char* S = alloc((size_t)padRows * DQKV * 2         // fused qkv
                  + (size_t)NTB * NHEAD * HDIM * LPAD * 2   // vt
                  + (size_t)ROWS_T * CCH * 2 + 4096);       // mix
  // temporal-phase views (dead outside temporal phases)
  h16*   a1  = (h16*)S;
  float* hh1 = (float*)(S + (((size_t)ROWS_S * CCH * 2 + 255) & ~(size_t)255));
  h16*   hh2 = (h16*)((char*)hh1 + (((size_t)ROWS_S * RDIM * 4 + 255) & ~(size_t)255));
  // attention-phase views
  h16* qkv = (h16*)S;
  h16* vt  = qkv + padRows * DQKV;
  h16* mix = vt + (size_t)NTB * NHEAD * HDIM * LPAD;
  // MLP-phase view
  h16* hmlp = (h16*)S;                               // ROWS_T x MMLP

  auto cast = [&](const float* src, h16* dst, long n) {
    castf2h_kernel<<<dim3((unsigned)((n + 255) / 256)), dim3(256), 0, stream>>>(src, dst, n);
  };
  cast(q_w, wqkv,                     (long)CCH * CCH);
  cast(k_w, wqkv + (size_t)CCH * CCH, (long)CCH * CCH);
  cast(v_w, wqkv + (size_t)2 * CCH * CCH, (long)CCH * CCH);
  concat3_kernel<<<dim3((DQKV + 255) / 256), dim3(256), 0, stream>>>(q_b, k_b, v_b, bqkv);
  cast(o_w, wo, (long)CCH * CCH);
  cast(fc1_w, wfc1, (long)MMLP * CCH);
  cast(fc2_w, wfc2, (long)CCH * MMLP);
  for (int i = 0; i < 2; ++i) {
    cast(tm_c1_w[i], wc1[i], (long)RDIM * CCH);
    cast(tm_c2_w[i], wc2[i], (long)CCH * RDIM);
  }

  dim3 blk128(128), blk256(256), blk32(32);
  dim3 gS_c1(ROWS_S / 128, RDIM / 64);    // 196 x 8
  dim3 gS_c2(ROWS_S / 128, CCH / 64);     // 196 x 12
  dim3 gT_768(ROWS_T / 128, CCH / 64);    // 197 x 12
  dim3 gT_qkv(ROWS_T / 128, DQKV / 64);   // 197 x 36
  dim3 gT_mlp(ROWS_T / 128, MMLP / 64);   // 197 x 48

  auto temporal = [&](int i, const float* xin, float* xout) {
    bnaff_kernel<<<dim3((ROWS_S * CCH + 255) / 256), blk256, 0, stream>>>(
        xin, tm_bn_g[i], tm_bn_b[i], tm_bn_rm[i], tm_bn_rv[i], a1);
    wmma_gemm<2><<<gS_c1, blk128, 0, stream>>>(a1, wc1[i], tm_c1_b[i],
        hh1, nullptr, nullptr, ROWS_S, RDIM, CCH);
    dwconv_kernel<<<dim3((ROWS_S * RDIM + 255) / 256), blk256, 0, stream>>>(
        hh1, tm_dw_w[i], tm_dw_b[i], hh2);
    wmma_gemm<3><<<gS_c2, blk128, 0, stream>>>(hh2, wc2[i], tm_c2_b[i],
        xout, nullptr, xin, ROWS_S, CCH, RDIM);
  };

  // ---- 1) temporal_mix #1 : x_work = x + temporal(x) ; CLS passthrough ----
  temporal(0, x, x_work);
  clscopy_kernel<<<dim3((NTB * CCH + 255) / 256), blk256, 0, stream>>>(x, x_work);

  // ---- 2) LN1 -> f16 ----
  layernorm_kernel<<<dim3(ROWS_T), blk256, 0, stream>>>(x_work, ln1_g, ln1_b, xn);

  // ---- 3) fused QKV projection (xn streamed once) ----
  wmma_gemm<0><<<gT_qkv, blk128, 0, stream>>>(xn, wqkv, bqkv, nullptr, qkv, nullptr,
                                              ROWS_T, DQKV, CCH);
  vtrans_kernel<<<dim3((NTB * NHEAD * HDIM * LPAD + 255) / 256), blk256, 0, stream>>>(qkv, vt);

  // ---- 4) fused attention -> mix (f16) ----
  attn_kernel<<<dim3(NTB, NHEAD, 13), blk32, 0, stream>>>(qkv, vt, mix);

  // ---- 5) O-proj + residual (in place on x_work) ----
  wmma_gemm<4><<<gT_768, blk128, 0, stream>>>(mix, wo, o_b, x_work, nullptr, x_work,
                                              ROWS_T, CCH, CCH);

  // ---- 6) temporal_mix #2 (in place) ----
  temporal(1, x_work, x_work);

  // ---- 7) LN2 -> f16 ----
  layernorm_kernel<<<dim3(ROWS_T), blk256, 0, stream>>>(x_work, ln2_g, ln2_b, xn);

  // ---- 8) MLP: fc1 + qgelu (f16), fc2 + residual -> d_out (f32) ----
  wmma_gemm<1><<<gT_mlp, blk128, 0, stream>>>(xn, wfc1, fc1_b, nullptr, hmlp, nullptr,
                                              ROWS_T, MMLP, CCH);
  wmma_gemm<4><<<gT_768, blk128, 0, stream>>>(hmlp, wfc2, fc2_b, out, nullptr, x_work,
                                              ROWS_T, CCH, MMLP);
}
